// SoftSOM_25443386261862
// MI455X (gfx1250) — matrix operand order repaired and checked by reading, not compiled, not run on hardware
//
#include <hip/hip_runtime.h>

typedef __bf16 bf16_t;
typedef __attribute__((ext_vector_type(16))) __bf16 v16bf;
typedef __attribute__((ext_vector_type(8)))  __bf16 v8bf;
typedef __attribute__((ext_vector_type(4)))  __bf16 v4bf;
typedef __attribute__((ext_vector_type(8)))  float  v8f;

#define N_TOK   131072
#define DIM     512
#define CELLS   256
#define WAVES   8
#define THREADS (WAVES * 32)
#define ROWS_PER_WG (WAVES * 16)
#define KSTEPS  (DIM / 32)     // 16 k-steps of 32 for the distance GEMMs
#define CTILES  (CELLS / 16)   // 16 column tiles of cells
#define NTILES  (DIM / 16)     // 32 output column tiles for blend GEMM
#define KCSTEPS (CELLS / 32)   // 8 k-steps for blend GEMM

// ---- LDS layout (bytes) ----
#define XS_STRIDE 520                        // 512 + 8 pad (halves)
#define XS_WAVE   (16 * XS_STRIDE)           // halves per wave
#define XS_BYTES  (WAVES * XS_WAVE * 2)      // 133120
#define SC_STRIDE 264                        // 256 + 8 pad (f32)
#define SC_WAVE   (16 * SC_STRIDE)
#define SC_BYTES  (WAVES * SC_WAVE * 4)      // 135168
#define FRAG_HALVES (KSTEPS * 32 * 16)       // 8192 halves = one B tile (16KB)
#define BSTAGE_BYTES (2 * FRAG_HALVES * 2)   // 32768 (Bp+Bg tile / 2x8KB ping-pong)
#define XN_BYTES  (WAVES * 16 * 4)           // 512
#define SMEM_BYTES (XS_BYTES + SC_BYTES + BSTAGE_BYTES + XN_BYTES)  // 301568
#define WB_STRIDE 264                        // bf16 weight row stride (256+8)
#define P3_HALVES (KCSTEPS * 32 * 16)        // 4096 halves = 8KB blend B tile
#define P3_BYTES  (P3_HALVES * 2)

// ---- workspace layout (bytes) ----
#define WS_BP   0
#define WS_BG   (WS_BP  + CTILES * FRAG_HALVES * 2)            // 262144
#define WS_BPT  (WS_BG  + CTILES * FRAG_HALVES * 2)            // 524288
#define WS_PN   (WS_BPT + NTILES * P3_HALVES * 2)              // 786432
#define WS_GN   (WS_PN  + CELLS * 4)
#define WS_GATE (WS_GN  + CELLS * 4)

// per-lane 16B async copy: global -> LDS, tracked by ASYNCcnt
__device__ __forceinline__ void async_b128(unsigned lds_off, const void* gaddr) {
  asm volatile("global_load_async_to_lds_b128 %0, %1, off"
               :: "v"(lds_off), "v"(gaddr)
               : "memory");
}
__device__ __forceinline__ void wait_async0() {
  asm volatile("s_wait_asynccnt 0" ::: "memory");
}

// Pre-swizzle P and G into WMMA B-fragment order, plus P^T fragments for the
// blend GEMM. Fragment storage: [tile][kstep][lane][16 halves], so each lane
// loads its 32-byte operand chunk contiguously.
__global__ void som_prep_frags(const float* __restrict__ P,
                               const float* __restrict__ G,
                               bf16_t* __restrict__ bp,
                               bf16_t* __restrict__ bg,
                               bf16_t* __restrict__ bpt) {
  int i = blockIdx.x * blockDim.x + threadIdx.x;
  const int FR = CTILES * FRAG_HALVES;   // 131072 (same count for bpt)
  if (i >= FR) return;
  int h = i & 15, L = (i >> 4) & 31;
  // distance-GEMM B fragments: B is [K=32][N=16]; lane = column, halves = K
  {
    int ks = (i >> 9) & 15, t = i >> 13;
    int cell = t * 16 + (L & 15);
    int k = ks * 32 + h + 16 * (L >> 4);
    bp[i] = (bf16_t)P[cell * DIM + k];
    bg[i] = (bf16_t)G[cell * DIM + k];
  }
  // blend-GEMM B fragments: K dim = cells, N dim = feature
  {
    int kc = (i >> 9) & 7, nt = i >> 12;
    int n = nt * 16 + (L & 15);
    int k = kc * 32 + h + 16 * (L >> 4);
    bpt[i] = (bf16_t)P[k * DIM + n];
  }
}

__global__ void som_prep_scalars(const float* __restrict__ P,
                                 const float* __restrict__ G,
                                 const float* __restrict__ gl,
                                 float* __restrict__ pn,
                                 float* __restrict__ gn,
                                 float* __restrict__ gate) {
  int c = threadIdx.x;                   // one block of 256
  float sp = 0.f, sg = 0.f;
  for (int d = 0; d < DIM; ++d) {
    float pv = (float)(bf16_t)P[c * DIM + d];   // bf16-rounded for consistency
    float gv = (float)(bf16_t)G[c * DIM + d];
    sp = fmaf(pv, pv, sp);
    sg = fmaf(gv, gv, sg);
  }
  pn[c] = sp;
  gn[c] = sg;
  gate[c] = 1.0f / (1.0f + __expf(-gl[c]));
}

__launch_bounds__(THREADS, 1)
__global__ void som_main(const float* __restrict__ x,
                         const bf16_t* __restrict__ bp,
                         const bf16_t* __restrict__ bg,
                         const bf16_t* __restrict__ bpt,
                         const float* __restrict__ pn,
                         const float* __restrict__ gn,
                         const float* __restrict__ gate,
                         const float* __restrict__ temp_raw,
                         float* __restrict__ out_blend,
                         float* __restrict__ out_w) {
  extern __shared__ char smem[];
  bf16_t* xs     = (bf16_t*)smem;
  float*  sc     = (float*)(smem + XS_BYTES);
  bf16_t* bstage = (bf16_t*)(smem + XS_BYTES + SC_BYTES);
  float*  xn     = (float*)(smem + XS_BYTES + SC_BYTES + BSTAGE_BYTES);

  const int tid  = threadIdx.x;
  const int wave = tid >> 5, lane = tid & 31;
  const int hi   = lane >> 4, lr = lane & 15;
  const long rowbase = (long)blockIdx.x * ROWS_PER_WG + wave * 16;
  // low 32 bits of a flat shared pointer == LDS byte offset (aperture in high bits)
  const unsigned bstage_lds = (unsigned)(uintptr_t)bstage;

  float invT;
  {
    float t = 1.0f / (1.0f + __expf(-temp_raw[0]));
    invT = 1.0f / (t * 0.999f + 0.001f);
  }

  // ---- phase 1: coalesced load of this wave's 16 x-rows -> bf16 LDS tile ----
  {
    const float4* x4 = (const float4*)(x + rowbase * DIM);
    bf16_t* xw = xs + wave * XS_WAVE;
#pragma unroll 4
    for (int it = 0; it < 64; ++it) {
      int e4 = it * 32 + lane;             // 2048 float4 per tile
      float4 v = x4[e4];
      int f = e4 * 4, r = f >> 9, c = f & 511;
      v4bf o;
      o.x = (bf16_t)v.x; o.y = (bf16_t)v.y; o.z = (bf16_t)v.z; o.w = (bf16_t)v.w;
      *(v4bf*)&xw[r * XS_STRIDE + c] = o;
    }
  }
  __syncthreads();

  // ---- phase 1.5: row norms of bf16-rounded x ----
  {
    const bf16_t* row = xs + wave * XS_WAVE + lr * XS_STRIDE + hi * 256;
    float a = 0.f;
#pragma unroll 8
    for (int i = 0; i < 256; ++i) {
      float v = (float)row[i];
      a = fmaf(v, v, a);
    }
    a += __shfl_xor(a, 16, 32);
    if (hi == 0) xn[wave * 16 + lr] = a;
  }

  // ---- phase 2: distances to all 256 cells, 16 cells per column tile ----
  for (int ct = 0; ct < CTILES; ++ct) {
    __syncthreads();                       // all waves done reading prev tile
    {
      const char* s0 = (const char*)(bp + ct * FRAG_HALVES);
      const char* s1 = (const char*)(bg + ct * FRAG_HALVES);
#pragma unroll
      for (int it = 0; it < 8; ++it) {     // 2048 x 16B = 32KB, async -> LDS
        int j = it * THREADS + tid;
        const char* src = (j < 1024) ? s0 + j * 16 : s1 + (j - 1024) * 16;
        async_b128(bstage_lds + j * 16, src);
      }
      if (ct + 1 < CTILES) {               // warm L2 for next tile
        __builtin_prefetch((const char*)(bp + (ct + 1) * FRAG_HALVES) + tid * 64, 0, 1);
        __builtin_prefetch((const char*)(bg + (ct + 1) * FRAG_HALVES) + tid * 64, 0, 1);
      }
      wait_async0();
    }
    __syncthreads();

    v8f accP = {0.f, 0.f, 0.f, 0.f, 0.f, 0.f, 0.f, 0.f};
    v8f accG = {0.f, 0.f, 0.f, 0.f, 0.f, 0.f, 0.f, 0.f};
    const bf16_t* arow_base = xs + wave * XS_WAVE + lr * XS_STRIDE + hi * 16;
#pragma unroll
    for (int ks = 0; ks < KSTEPS; ++ks) {
      union { v16bf v; v8bf h[2]; } a;
      const bf16_t* ar = arow_base + ks * 32;
      a.h[0] = *(const v8bf*)ar;           // K = 32ks + 16hi + [0..7]
      a.h[1] = *(const v8bf*)(ar + 8);     // K = 32ks + 16hi + [8..15]
      v16bf bpv = *(const v16bf*)&bstage[(ks * 32 + lane) * 16];
      v16bf bgv = *(const v16bf*)&bstage[FRAG_HALVES + (ks * 32 + lane) * 16];
      accP = __builtin_amdgcn_wmma_f32_16x16x32_bf16(false, a.v, false, bpv,
                                                     (short)0, accP, false, false);
      accG = __builtin_amdgcn_wmma_f32_16x16x32_bf16(false, a.v, false, bgv,
                                                     (short)0, accG, false, false);
    }
    int col = ct * 16 + lr;
    float pnc = pn[col], gnc = gn[col];
    float* scw = sc + wave * SC_WAVE;
#pragma unroll
    for (int v = 0; v < 8; ++v) {
      int r = v + 8 * hi;
      float xnr = xn[wave * 16 + r];
      float d2p = fmaxf(xnr + pnc - 2.0f * accP[v], 0.0f);
      float d2g = fmaxf(xnr + gnc - 2.0f * accG[v], 0.0f);
      scw[r * SC_STRIDE + col] = -(sqrtf(d2p) + sqrtf(d2g)) * invT;
    }
  }
  __syncthreads();                         // all phase-2 bstage reads done

  // ---- phase 2.5: fused softmax * gate, renorm; write weights; stash bf16 A ----
  {
    float* srow = sc + wave * SC_WAVE + lr * SC_STRIDE;
    int cb = hi * 128;
    float m = -3.0e38f;
#pragma unroll 4
    for (int i = 0; i < 128; ++i) m = fmaxf(m, srow[cb + i]);
    m = fmaxf(m, __shfl_xor(m, 16, 32));
    float Z = 0.f, S = 0.f;
#pragma unroll 4
    for (int i = 0; i < 128; ++i) {
      int c = cb + i;
      float e = __expf(srow[c] - m);
      float w = e * gate[c];
      Z += e; S += w;
      srow[c] = w;
    }
    Z += __shfl_xor(Z, 16, 32);
    S += __shfl_xor(S, 16, 32);
    // softmax*gate renormalized: w / (sum(w) + 1e-8*sum(e))
    float inv = 1.0f / (S + 1e-8f * Z);
    bf16_t* wrow = xs + wave * XS_WAVE + lr * WB_STRIDE;   // reuse x region
    float4* ow = (float4*)(out_w + (rowbase + lr) * CELLS);
#pragma unroll 4
    for (int i4 = 0; i4 < 32; ++i4) {
      int c = cb + i4 * 4;
      float w0 = srow[c + 0] * inv, w1 = srow[c + 1] * inv;
      float w2 = srow[c + 2] * inv, w3 = srow[c + 3] * inv;
      v4bf o;
      o.x = (bf16_t)w0; o.y = (bf16_t)w1; o.z = (bf16_t)w2; o.w = (bf16_t)w3;
      *(v4bf*)&wrow[c] = o;
      float4 f4; f4.x = w0; f4.y = w1; f4.z = w2; f4.w = w3;
      ow[c >> 2] = f4;
    }
  }

  // ---- phase 3: blended = weights @ prototypes, double-buffered async B tiles ----
  {
#pragma unroll
    for (int it = 0; it < 2; ++it) {       // prologue: tile 0 -> buffer 0
      int j = it * THREADS + tid;
      async_b128(bstage_lds + j * 16, (const char*)bpt + j * 16);
    }
    wait_async0();
  }
  __syncthreads();

  for (int nt = 0; nt < NTILES; ++nt) {
    if (nt + 1 < NTILES) {                 // prefetch next tile into other buffer
      const char* src = (const char*)(bpt + (nt + 1) * P3_HALVES);
      unsigned db = bstage_lds + ((nt + 1) & 1) * P3_BYTES;
#pragma unroll
      for (int it = 0; it < 2; ++it) {
        int j = it * THREADS + tid;
        async_b128(db + j * 16, src + j * 16);
      }
    }

    v8f acc = {0.f, 0.f, 0.f, 0.f, 0.f, 0.f, 0.f, 0.f};
    const bf16_t* bbase = bstage + (nt & 1) * P3_HALVES;
    const bf16_t* arow = xs + wave * XS_WAVE + lr * WB_STRIDE + hi * 16;
#pragma unroll
    for (int kc = 0; kc < KCSTEPS; ++kc) {
      union { v16bf v; v8bf h[2]; } a;
      const bf16_t* ap = arow + kc * 32;
      a.h[0] = *(const v8bf*)ap;
      a.h[1] = *(const v8bf*)(ap + 8);
      v16bf b = *(const v16bf*)&bbase[(kc * 32 + lane) * 16];
      acc = __builtin_amdgcn_wmma_f32_16x16x32_bf16(false, a.v, false, b,
                                                    (short)0, acc, false, false);
    }
#pragma unroll
    for (int v = 0; v < 8; ++v)
      out_blend[(rowbase + v + 8 * hi) * DIM + nt * 16 + lr] = acc[v];

    wait_async0();                         // next buffer filled ...
    __syncthreads();                       // ... and visible to all waves
  }
}

extern "C" void kernel_launch(void* const* d_in, const int* in_sizes, int n_in,
                              void* d_out, int out_size, void* d_ws, size_t ws_size,
                              hipStream_t stream) {
  const float* x  = (const float*)d_in[0];
  const float* P  = (const float*)d_in[1];
  const float* G  = (const float*)d_in[2];
  const float* tr = (const float*)d_in[3];
  const float* gl = (const float*)d_in[4];
  float* out = (float*)d_out;

  char* ws = (char*)d_ws;
  bf16_t* bpw  = (bf16_t*)(ws + WS_BP);
  bf16_t* bgw  = (bf16_t*)(ws + WS_BG);
  bf16_t* bptw = (bf16_t*)(ws + WS_BPT);
  float*  pnw  = (float*)(ws + WS_PN);
  float*  gnw  = (float*)(ws + WS_GN);
  float*  gatw = (float*)(ws + WS_GATE);

  som_prep_frags<<<512, 256, 0, stream>>>(P, G, bpw, bgw, bptw);
  som_prep_scalars<<<1, 256, 0, stream>>>(P, G, gl, pnw, gnw, gatw);

  (void)hipFuncSetAttribute((const void*)som_main,
                            hipFuncAttributeMaxDynamicSharedMemorySize, SMEM_BYTES);
  som_main<<<N_TOK / ROWS_PER_WG, THREADS, SMEM_BYTES, stream>>>(
      x, bpw, bgw, bptw, pnw, gnw, gatw, tr,
      out, out + (size_t)N_TOK * DIM);
}
